// GNN_Model_19842748907774
// MI455X (gfx1250) — matrix operand order repaired and checked by reading, not compiled, not run on hardware
//
#include <hip/hip_runtime.h>
#include <hip/hip_bf16.h>
#include <stdint.h>

// ---------------------------------------------------------------------------
// GATv2 x4 + mean-pool + linear for MI455X (gfx1250, wave32, WMMA).
// Dense transforms: v_wmma_f32_16x16x32_f16, K templated (no guards),
// 4 col-tiles per wave (A-frag reuse). Edge pipeline templated on (H,C),
// float4 vectorized; xl/xr L2-resident; segment max via ordered-uint
// atomicMax; segment sums via global_atomic_add_f32.
// ---------------------------------------------------------------------------

typedef __attribute__((ext_vector_type(16))) _Float16 v16h;
typedef __attribute__((ext_vector_type(8)))  float    v8f;

#define NN   60000
#define EE   960000
#define GG   256
#define ETOT (EE + NN)

// ---- float <-> ordered-uint (for float atomic max via uint atomicMax) -----
__device__ __forceinline__ unsigned f2ord(float f) {
  unsigned u = __float_as_uint(f);
  return (u & 0x80000000u) ? ~u : (u | 0x80000000u);
}
__device__ __forceinline__ float ord2f(unsigned o) {
  return (o & 0x80000000u) ? __uint_as_float(o & 0x7FFFFFFFu)
                           : __uint_as_float(~o);
}
#define ORD_NEG_INF 0x007FFFFFu   // f2ord(-inf)

// ---------------------------------------------------------------------------
__global__ void k_fill_u32(unsigned* __restrict__ p, unsigned v, long long n) {
  for (long long i = blockIdx.x * (long long)blockDim.x + threadIdx.x; i < n;
       i += (long long)gridDim.x * blockDim.x)
    p[i] = v;
}

// Zero-pad x (N x 5) -> xpad (N x 32)
__global__ void k_pad_x(const float* __restrict__ x, float* __restrict__ xp) {
  for (long long i = blockIdx.x * (long long)blockDim.x + threadIdx.x;
       i < (long long)NN * 32; i += (long long)gridDim.x * blockDim.x) {
    int c = (int)(i & 31);
    int n = (int)(i >> 5);
    xp[i] = (c < 5) ? x[(size_t)n * 5 + c] : 0.f;
  }
}

// ---------------------------------------------------------------------------
// Pre-swizzle f32 weight W[K x outF] into f16 B-fragments (16-bit B 32x16
// layout: g=lane>>4; j 0..7 -> K=8g+j, j 8..15 -> K=16+8g+(j-8);
// lane&15 = column). bfrag[((ct*KT + kt)*32 + lane)*16 + j]
// ---------------------------------------------------------------------------
__global__ void k_prep_bfrag(const float* __restrict__ W, int K, int Kpad,
                             int outF, _Float16* __restrict__ bfrag) {
  int KT = Kpad >> 5;
  int CT = outF >> 4;
  long long total = (long long)CT * KT * 32 * 16;
  for (long long idx = blockIdx.x * (long long)blockDim.x + threadIdx.x;
       idx < total; idx += (long long)gridDim.x * blockDim.x) {
    int j    = (int)(idx & 15);
    int lane = (int)((idx >> 4) & 31);
    int t    = (int)(idx >> 9);          // ct*KT + kt
    int kt = t % KT, ct = t / KT;
    int v = j >> 1, pbit = j & 1, g = lane >> 4, n = lane & 15;
    int k = kt * 32 + 8 * g + j + ((j >> 3) << 3);
    float w = (k < K) ? W[(size_t)k * outF + ct * 16 + n] : 0.f;
    bfrag[idx] = (_Float16)w;
  }
}

// ---------------------------------------------------------------------------
// WMMA GEMM, K compile-time (stride == K, multiple of 32, no guards).
// Grid: (NN/16, outF/64, 2{L,R}); block 32 (one wave). Each wave computes
// a 16x64 output strip: A-frag loaded once per K-step, 4 WMMAs.
// A layout (16-bit A 16x32): lane m=lane&15 row; g=lane>>4:
//   j 0..7  -> k = kt*32 + 8g + j
//   j 8..15 -> k = kt*32 + 16 + 8g + (j-8)
// ---------------------------------------------------------------------------
template <int K>
__global__ void k_gemm_wmma(const float* __restrict__ A,
                            const _Float16* __restrict__ bfragL,
                            const _Float16* __restrict__ bfragR,
                            const float* __restrict__ biasL,
                            const float* __restrict__ biasR,
                            float* __restrict__ outL, float* __restrict__ outR,
                            int outF) {
  const int lane = threadIdx.x;
  const int rt = blockIdx.x;
  const int ct0 = blockIdx.y * 4;
  const bool isR = (blockIdx.z != 0);
  const _Float16* __restrict__ bfrag = isR ? bfragR : bfragL;
  const float* __restrict__ bias = isR ? biasR : biasL;
  float* __restrict__ out = isR ? outR : outL;

  const int m = lane & 15;
  const int g = lane >> 4;
  constexpr int KT = K / 32;
  const float4* __restrict__ arow =
      (const float4*)(A + (size_t)(rt * 16 + m) * K);

  v8f acc0 = {}, acc1 = {}, acc2 = {}, acc3 = {};
#pragma unroll
  for (int kt = 0; kt < KT; ++kt) {
    const float4 a0 = arow[8 * kt + 2 * g];
    const float4 a1 = arow[8 * kt + 2 * g + 1];
    const float4 a2 = arow[8 * kt + 4 + 2 * g];
    const float4 a3 = arow[8 * kt + 4 + 2 * g + 1];
    v16h a;
    a[0] = (_Float16)a0.x;  a[1] = (_Float16)a0.y;
    a[2] = (_Float16)a0.z;  a[3] = (_Float16)a0.w;
    a[4] = (_Float16)a1.x;  a[5] = (_Float16)a1.y;
    a[6] = (_Float16)a1.z;  a[7] = (_Float16)a1.w;
    a[8] = (_Float16)a2.x;  a[9] = (_Float16)a2.y;
    a[10] = (_Float16)a2.z; a[11] = (_Float16)a2.w;
    a[12] = (_Float16)a3.x; a[13] = (_Float16)a3.y;
    a[14] = (_Float16)a3.z; a[15] = (_Float16)a3.w;

    const _Float16* bbase = bfrag + ((size_t)(ct0 * KT + kt) * 32 + lane) * 16;
    const size_t bstride = (size_t)KT * 32 * 16;
    v16h b0 = *(const v16h*)(bbase);
    v16h b1 = *(const v16h*)(bbase + bstride);
    v16h b2 = *(const v16h*)(bbase + 2 * bstride);
    v16h b3 = *(const v16h*)(bbase + 3 * bstride);
    acc0 = __builtin_amdgcn_wmma_f32_16x16x32_f16(false, a, false, b0,
                                                  (short)0, acc0, false, false);
    acc1 = __builtin_amdgcn_wmma_f32_16x16x32_f16(false, a, false, b1,
                                                  (short)0, acc1, false, false);
    acc2 = __builtin_amdgcn_wmma_f32_16x16x32_f16(false, a, false, b2,
                                                  (short)0, acc2, false, false);
    acc3 = __builtin_amdgcn_wmma_f32_16x16x32_f16(false, a, false, b3,
                                                  (short)0, acc3, false, false);
  }

  // C/D layout: lane n=lane&15 column; lanes 16-31 hold rows M=8..15.
  const int rowBase = rt * 16 + 8 * g;
#pragma unroll
  for (int c = 0; c < 4; ++c) {
    const v8f acc = (c == 0) ? acc0 : (c == 1) ? acc1 : (c == 2) ? acc2 : acc3;
    const int n = (ct0 + c) * 16 + m;
    const float bv = bias[n];
#pragma unroll
    for (int r = 0; r < 8; ++r)
      out[(size_t)(rowBase + r) * outF + n] = acc[r] + bv;
  }
}

// ---------------------------------------------------------------------------
// Self-loop attr: cnt[dst]++, loop_sum[dst] += edge_features[e]
// ---------------------------------------------------------------------------
__global__ void k_loop_scatter(const int* __restrict__ ei,
                               const float* __restrict__ ef,
                               float* __restrict__ cnt,
                               float* __restrict__ lsum) {
  for (long long e = blockIdx.x * (long long)blockDim.x + threadIdx.x; e < EE;
       e += (long long)gridDim.x * blockDim.x) {
    int dst = ei[EE + e];
    atomicAdd(cnt + dst, 1.0f);
    atomicAdd(lsum + (size_t)dst * 3 + 0, ef[e * 3 + 0]);
    atomicAdd(lsum + (size_t)dst * 3 + 1, ef[e * 3 + 1]);
    atomicAdd(lsum + (size_t)dst * 3 + 2, ef[e * 3 + 2]);
  }
}
__global__ void k_loop_final(const float* __restrict__ cnt,
                             const float* __restrict__ lsum,
                             float* __restrict__ la) {
  for (long long i = blockIdx.x * (long long)blockDim.x + threadIdx.x;
       i < (long long)NN * 3; i += (long long)gridDim.x * blockDim.x) {
    int nidx = (int)(i / 3);
    float c = cnt[nidx];
    la[i] = lsum[i] / (c > 1.0f ? c : 1.0f);
  }
}

// ---------------------------------------------------------------------------
// Pass 1: attention logits + segment max. One thread per (edge, head).
// ---------------------------------------------------------------------------
template <int H, int C>
__global__ void k_edge_logit(const int* __restrict__ ei,
                             const float* __restrict__ ef,
                             const float* __restrict__ la,
                             const float* __restrict__ xl,
                             const float* __restrict__ xr,
                             const float* __restrict__ We,
                             const float* __restrict__ att,
                             float* __restrict__ logit,
                             unsigned* __restrict__ mord) {
  constexpr int F = H * C;
  const long long total = (long long)ETOT * H;
  for (long long t = blockIdx.x * (long long)blockDim.x + threadIdx.x;
       t < total; t += (long long)gridDim.x * blockDim.x) {
    const int h = (int)(t & (H - 1));
    const long long e = t >> (H == 4 ? 2 : 1);
    int src, dst;
    const float* ea;
    if (e < EE) { src = ei[e]; dst = ei[EE + e]; ea = ef + e * 3; }
    else        { int nn2 = (int)(e - EE); src = dst = nn2; ea = la + (size_t)nn2 * 3; }
    const float e0 = ea[0], e1 = ea[1], e2 = ea[2];
    const float* __restrict__ pl = xl + (size_t)src * F + h * C;
    const float* __restrict__ pr = xr + (size_t)dst * F + h * C;
    const float* __restrict__ w0 = We + h * C;
    const float* __restrict__ pat = att + h * C;
    float acc = 0.f;
#pragma unroll
    for (int c = 0; c < C; c += 4) {
      const float4 l4 = *(const float4*)(pl + c);
      const float4 r4 = *(const float4*)(pr + c);
      const float4 wa = *(const float4*)(w0 + c);
      const float4 wb = *(const float4*)(w0 + F + c);
      const float4 wc = *(const float4*)(w0 + 2 * F + c);
      const float4 at = *(const float4*)(pat + c);
#define GAT_COMP(q)                                           \
      {                                                       \
        float s = l4.q + r4.q + e0 * wa.q + e1 * wb.q + e2 * wc.q; \
        s = (s >= 0.f) ? s : 0.2f * s;                        \
        acc += s * at.q;                                      \
      }
      GAT_COMP(x) GAT_COMP(y) GAT_COMP(z) GAT_COMP(w)
#undef GAT_COMP
    }
    logit[e * H + h] = acc;
    atomicMax(mord + (size_t)dst * H + h, f2ord(acc));
  }
}

// Pass 2: p = exp(logit - m[dst]); den[dst] += p (in place over logit)
template <int H>
__global__ void k_edge_exp(const int* __restrict__ ei,
                           float* __restrict__ logit,
                           const unsigned* __restrict__ mord,
                           float* __restrict__ den) {
  const long long total = (long long)ETOT * H;
  for (long long t = blockIdx.x * (long long)blockDim.x + threadIdx.x;
       t < total; t += (long long)gridDim.x * blockDim.x) {
    const int h = (int)(t & (H - 1));
    const long long e = t >> (H == 4 ? 2 : 1);
    const int dst = (e < EE) ? ei[EE + e] : (int)(e - EE);
    const float m = ord2f(mord[(size_t)dst * H + h]);
    const float p = __expf(logit[t] - m);
    logit[t] = p;
    atomicAdd(den + (size_t)dst * H + h, p);
  }
}

// Pass 2b: alpha = p / den[dst]  (in place)
template <int H>
__global__ void k_edge_alpha(const int* __restrict__ ei,
                             float* __restrict__ logit,
                             const float* __restrict__ den) {
  const long long total = (long long)ETOT * H;
  for (long long t = blockIdx.x * (long long)blockDim.x + threadIdx.x;
       t < total; t += (long long)gridDim.x * blockDim.x) {
    const int h = (int)(t & (H - 1));
    const long long e = t >> (H == 4 ? 2 : 1);
    const int dst = (e < EE) ? ei[EE + e] : (int)(e - EE);
    logit[t] = logit[t] / den[(size_t)dst * H + h];
  }
}

// Pass 3: out[dst] += xl[src] * alpha; one thread per (edge, 4 channels).
template <int H, int C>
__global__ void k_edge_aggr(const int* __restrict__ ei,
                            const float* __restrict__ xl,
                            const float* __restrict__ alpha,
                            float* __restrict__ out) {
  constexpr int F = H * C;
  constexpr int F4 = F / 4;
  const long long total = (long long)ETOT * F4;
  for (long long t = blockIdx.x * (long long)blockDim.x + threadIdx.x;
       t < total; t += (long long)gridDim.x * blockDim.x) {
    const int ch4 = (int)(t % F4);
    const long long e = t / F4;
    const int h = (ch4 * 4) / C;
    int src, dst;
    if (e < EE) { src = ei[e]; dst = ei[EE + e]; }
    else        { src = dst = (int)(e - EE); }
    const float a = alpha[e * H + h];
    const float4 v = *(const float4*)(xl + (size_t)src * F + ch4 * 4);
    float* o = out + (size_t)dst * F + ch4 * 4;
    atomicAdd(o + 0, v.x * a);
    atomicAdd(o + 1, v.y * a);
    atomicAdd(o + 2, v.z * a);
    atomicAdd(o + 3, v.w * a);
  }
}

// h = relu(out + bias) (in place)
__global__ void k_bias_relu(float* __restrict__ buf,
                            const float* __restrict__ b, int F) {
  const long long total = (long long)NN * F;
  for (long long i = blockIdx.x * (long long)blockDim.x + threadIdx.x;
       i < total; i += (long long)gridDim.x * blockDim.x) {
    float v = buf[i] + b[(int)(i % F)];
    buf[i] = v > 0.f ? v : 0.f;
  }
}

// Pooling accumulation: psum[batch[n]] += h[n]; pcnt[batch[n]] += 1
__global__ void k_pool_acc(const float* __restrict__ h,
                           const int* __restrict__ batch,
                           float* __restrict__ psum,
                           float* __restrict__ pcnt) {
  const long long total = (long long)NN * 16;  // 16 float4 per node (64 ch)
  for (long long t = blockIdx.x * (long long)blockDim.x + threadIdx.x;
       t < total; t += (long long)gridDim.x * blockDim.x) {
    const int c4 = (int)(t & 15);
    const int n = (int)(t >> 4);
    const int g = batch[n];
    const float4 v = *(const float4*)(h + (size_t)n * 64 + c4 * 4);
    float* o = psum + (size_t)g * 64 + c4 * 4;
    atomicAdd(o + 0, v.x);
    atomicAdd(o + 1, v.y);
    atomicAdd(o + 2, v.z);
    atomicAdd(o + 3, v.w);
    if (c4 == 0) atomicAdd(pcnt + g, 1.0f);
  }
}

// Final: out[g,o] = blin[o] + (psum[g]/max(cnt,1)) . wlin[:,o]
__global__ void k_final_lin(const float* __restrict__ psum,
                            const float* __restrict__ pcnt,
                            const float* __restrict__ wlin,
                            const float* __restrict__ blin,
                            float* __restrict__ out) {
  for (int t = blockIdx.x * blockDim.x + threadIdx.x; t < GG * 32;
       t += gridDim.x * blockDim.x) {
    const int o = t & 31;
    const int g = t >> 5;
    const float c = pcnt[g];
    const float inv = 1.0f / (c > 1.0f ? c : 1.0f);
    float acc = blin[o];
#pragma unroll 4
    for (int k = 0; k < 64; ++k)
      acc += psum[(size_t)g * 64 + k] * inv * wlin[k * 32 + o];
    out[t] = acc;
  }
}

// ---------------------------------------------------------------------------
struct Lay {
  const float *wl, *bl, *wr, *br, *we, *att, *cb;
  int K, H, C;
};

static const dim3 B256(256);
static const dim3 Gbig(4096);
static const dim3 Gmed(512);

template <int K, int H, int C>
static void run_layer(const float* hin, float* hout, const Lay& L,
                      const _Float16* bfL, const _Float16* bfR, float* xl,
                      float* xr, float* logit, unsigned* mord, float* den,
                      const int* ei, const float* ef, const float* la,
                      hipStream_t stream) {
  constexpr int F = H * C;
  dim3 ggrid(NN / 16, F / 64, 2);
  k_gemm_wmma<K><<<ggrid, dim3(32), 0, stream>>>(hin, bfL, bfR, L.bl, L.br, xl,
                                                 xr, F);
  k_fill_u32<<<Gbig, B256, 0, stream>>>(mord, ORD_NEG_INF, (long long)NN * H);
  k_edge_logit<H, C><<<Gbig, B256, 0, stream>>>(ei, ef, la, xl, xr, L.we,
                                                L.att, logit, mord);
  k_fill_u32<<<Gbig, B256, 0, stream>>>((unsigned*)den, 0, (long long)NN * H);
  k_fill_u32<<<Gbig, B256, 0, stream>>>((unsigned*)hout, 0, (long long)NN * F);
  k_edge_exp<H><<<Gbig, B256, 0, stream>>>(ei, logit, mord, den);
  k_edge_alpha<H><<<Gbig, B256, 0, stream>>>(ei, logit, den);
  k_edge_aggr<H, C><<<Gbig, B256, 0, stream>>>(ei, xl, logit, hout);
  k_bias_relu<<<Gbig, B256, 0, stream>>>(hout, L.cb, F);
}

extern "C" void kernel_launch(void* const* d_in, const int* in_sizes, int n_in,
                              void* d_out, int out_size, void* d_ws,
                              size_t ws_size, hipStream_t stream) {
  (void)in_sizes; (void)n_in; (void)out_size; (void)ws_size;

  const float* x   = (const float*)d_in[0];
  const int*   ei  = (const int*)d_in[1];
  const float* ef  = (const float*)d_in[2];
  const int*   bat = (const int*)d_in[3];
  static const int Ks[4] = {5, 64, 128, 128};
  static const int Hs[4] = {4, 4, 4, 2};
  static const int Cs[4] = {16, 32, 32, 32};
  Lay L[4];
  for (int i = 0; i < 4; ++i) {
    const int b = 4 + i * 7;
    L[i] = { (const float*)d_in[b + 0], (const float*)d_in[b + 1],
             (const float*)d_in[b + 2], (const float*)d_in[b + 3],
             (const float*)d_in[b + 4], (const float*)d_in[b + 5],
             (const float*)d_in[b + 6], Ks[i], Hs[i], Cs[i] };
  }
  const float* wlin = (const float*)d_in[32];
  const float* blin = (const float*)d_in[33];
  float* out = (float*)d_out;

  // ---- workspace carve (256B aligned regions) ----
  char* wp = (char*)d_ws;
  auto carve = [&](size_t bytes) -> char* {
    char* r = wp;
    wp += (bytes + 255) & ~(size_t)255;
    return r;
  };
  float* hA    = (float*)carve((size_t)NN * 128 * 4);
  float* hB    = (float*)carve((size_t)NN * 128 * 4);
  float* xl    = (float*)carve((size_t)NN * 128 * 4);
  float* xr    = (float*)carve((size_t)NN * 128 * 4);
  float* xpad  = (float*)carve((size_t)NN * 32 * 4);
  float* logit = (float*)carve((size_t)ETOT * 4 * 4);
  unsigned* mord = (unsigned*)carve((size_t)NN * 4 * 4);
  float* den   = (float*)carve((size_t)NN * 4 * 4);
  float* lsum  = (float*)carve((size_t)NN * 3 * 4);
  float* cnt   = (float*)carve((size_t)NN * 4);
  float* la    = (float*)carve((size_t)NN * 3 * 4);
  float* psum  = (float*)carve((size_t)GG * 64 * 4);
  float* pcnt  = (float*)carve((size_t)GG * 4);
  _Float16* bfL[4]; _Float16* bfR[4];
  for (int i = 0; i < 4; ++i) {
    int Kpad = (L[i].K + 31) & ~31;
    int F = L[i].H * L[i].C;
    size_t frag = (size_t)(Kpad >> 5) * (F >> 4) * 512;  // f16 elements
    bfL[i] = (_Float16*)carve(frag * 2);
    bfR[i] = (_Float16*)carve(frag * 2);
  }

  // ---- weight fragments + padded x (cheap pre-passes) ----
  for (int i = 0; i < 4; ++i) {
    int Kpad = (L[i].K + 31) & ~31;
    int F = L[i].H * L[i].C;
    k_prep_bfrag<<<Gmed, B256, 0, stream>>>(L[i].wl, L[i].K, Kpad, F, bfL[i]);
    k_prep_bfrag<<<Gmed, B256, 0, stream>>>(L[i].wr, L[i].K, Kpad, F, bfR[i]);
  }
  k_pad_x<<<Gbig, B256, 0, stream>>>(x, xpad);

  // ---- self-loop edge attributes (mean of incoming) ----
  k_fill_u32<<<Gbig, B256, 0, stream>>>((unsigned*)cnt, 0, NN);
  k_fill_u32<<<Gbig, B256, 0, stream>>>((unsigned*)lsum, 0, (long long)NN * 3);
  k_loop_scatter<<<Gbig, B256, 0, stream>>>(ei, ef, cnt, lsum);
  k_loop_final<<<Gbig, B256, 0, stream>>>(cnt, lsum, la);

  // ---- 4 GATv2 layers (K = padded A stride; layer1 uses xpad) ----
  run_layer<32, 4, 16>(xpad, hA, L[0], bfL[0], bfR[0], xl, xr, logit, mord,
                       den, ei, ef, la, stream);
  run_layer<64, 4, 32>(hA, hB, L[1], bfL[1], bfR[1], xl, xr, logit, mord, den,
                       ei, ef, la, stream);
  run_layer<128, 4, 32>(hB, hA, L[2], bfL[2], bfR[2], xl, xr, logit, mord, den,
                        ei, ef, la, stream);
  run_layer<128, 2, 32>(hA, hB, L[3], bfL[3], bfR[3], xl, xr, logit, mord, den,
                        ei, ef, la, stream);

  // ---- global mean pool + final linear ----
  k_fill_u32<<<Gbig, B256, 0, stream>>>((unsigned*)psum, 0, (long long)GG * 64);
  k_fill_u32<<<Gbig, B256, 0, stream>>>((unsigned*)pcnt, 0, GG);
  k_pool_acc<<<Gbig, B256, 0, stream>>>(hB, bat, psum, pcnt);
  k_final_lin<<<dim3(32), B256, 0, stream>>>(psum, pcnt, wlin, blin, out);
}
